// Elmo_LSTM_BPE_35416300323643
// MI455X (gfx1250) — compile-verified, matrix operand order checked by reading, not run on hardware
//
#include <hip/hip_runtime.h>
#include <hip/hip_bf16.h>
#include <stdint.h>

// ---------------------------------------------------------------------------
// ELMo bi-LSTM w/ projection for MI455X (gfx1250, wave32, WMMA bf16 path)
// B=32 T=128 D=H=512 C=4096 G=4C=16384 L=2, 2 directions
// ---------------------------------------------------------------------------

typedef __bf16 bf16;
typedef __attribute__((ext_vector_type(16))) __bf16 v16bf;
typedef __attribute__((ext_vector_type(8)))  float   v8f;
typedef unsigned int u32;
typedef u32 u32x4 __attribute__((ext_vector_type(4)));
typedef int i32x4 __attribute__((ext_vector_type(4)));
typedef int i32x8 __attribute__((ext_vector_type(8)));

#define B_ 32
#define T_ 128
#define D_ 512
#define H_ 512
#define C_ 4096
#define G_ 16384
#define L_ 2

#define WMMA_BF16(A, Bm, Cm) \
  __builtin_amdgcn_wmma_f32_16x16x32_bf16(false, (A), false, (Bm), (short)0, (Cm), false, false)

__device__ __forceinline__ bf16 f2b(float f) { return (bf16)f; }

union TileU { v16bf v; uint4 q[2]; };

// A/B operand tile (16x32 bf16) per ISA layout:
// lanes 0-15: row=lane, K-chunks [k0..k0+7] and [k0+16..k0+23]
// lanes 16-31: row=lane-16, K-chunks [k0+8..k0+15] and [k0+24..k0+31]
__device__ __forceinline__ v16bf load_tile(const bf16* base, int ld,
                                           int row0, int k0, int l) {
  const bf16* p = base + (size_t)(row0 + (l & 15)) * ld + k0 + ((l >> 4) << 3);
  TileU u;
  u.q[0] = *(const uint4*)(p);
  u.q[1] = *(const uint4*)(p + 16);
  return u.v;
}

// A operand for the input GEMM: GEMM row m = t*32+b maps to x[b][t][*]
__device__ __forceinline__ v16bf load_tile_x(const bf16* __restrict__ x,
                                             int m0, int k0, int l) {
  int m = m0 + (l & 15);
  int t = m >> 5, b = m & 31;
  const bf16* p = x + ((size_t)b * T_ + t) * D_ + k0 + ((l >> 4) << 3);
  TileU u;
  u.q[0] = *(const uint4*)(p);
  u.q[1] = *(const uint4*)(p + 16);
  return u.v;
}

// C/D 16x16 f32 tile: lane n = col0+(l&15); VGPR r holds row m0 + r + 8*(l>=16)
__device__ __forceinline__ v8f load_ctile(const float* __restrict__ base, int ld,
                                          int row0, int col0, int l) {
  int n = col0 + (l & 15);
  int mo = row0 + ((l >> 4) << 3);
  v8f c;
#pragma unroll
  for (int r = 0; r < 8; r++) c[r] = base[(size_t)(mo + r) * ld + n];
  return c;
}

__device__ __forceinline__ void store_ctile(float* __restrict__ base, int ld,
                                            int row0, int col0, v8f c, int l) {
  int n = col0 + (l & 15);
  int mo = row0 + ((l >> 4) << 3);
#pragma unroll
  for (int r = 0; r < 8; r++) base[(size_t)(mo + r) * ld + n] = c[r];
}

// --------------------- TDM: stage 16KB contiguous tile into LDS ------------
// Flat copy as a 1-row tensor tile: data_size=8B, tile_dim0=2048 elements.
__device__ __forceinline__ void stage_16k_to_lds(const bf16* gsrc, bf16* smem) {
#if __has_builtin(__builtin_amdgcn_tensor_load_to_lds)
  if ((threadIdx.x >> 5) == 0) {  // one wave issues the TDM op (EXEC ignored)
    unsigned long long ga = (unsigned long long)gsrc;
    u32 lds_off = (u32)(size_t)smem;  // generic->LDS byte offset (low 32 bits)
    u32x4 g0;
    g0[0] = 1u;                                            // count=1
    g0[1] = lds_off;                                       // lds_addr
    g0[2] = (u32)(ga & 0xffffffffu);                       // global_addr[31:0]
    g0[3] = (u32)((ga >> 32) & 0x1ffffffu) | (2u << 30);   // addr[56:32]|type=2
    i32x8 g1;
    g1[0] = (3 << 16);            // workgroup_mask=0, data_size=3 (8 bytes)
    g1[1] = (int)((2048u & 0xffffu) << 16);  // tensor_dim0[15:0]=2048
    g1[2] = (int)(1u << 16);      // tensor_dim0[31:16]=0, tensor_dim1[15:0]=1
    g1[3] = (int)(2048u << 16);   // tensor_dim1[31:16]=0, tile_dim0=2048
    g1[4] = 1;                    // tile_dim1=1, tile_dim2=0
    g1[5] = 2048;                 // tensor_dim0_stride[31:0]
    g1[6] = 0;
    g1[7] = 0;
    i32x4 z4 = {0, 0, 0, 0};
#if __clang_major__ >= 23
    i32x8 z8 = {0, 0, 0, 0, 0, 0, 0, 0};
    __builtin_amdgcn_tensor_load_to_lds(g0, g1, z4, z4, z8, 0);
#else
    __builtin_amdgcn_tensor_load_to_lds(g0, g1, z4, z4, 0);
#endif
#if __has_builtin(__builtin_amdgcn_s_wait_tensorcnt)
    __builtin_amdgcn_s_wait_tensorcnt(0);
#else
    asm volatile("s_wait_tensorcnt 0x0" ::: "memory");
#endif
  }
#else
  // Fallback: cooperative global->LDS staging (ds_store_b128 path)
  const uint4* src = (const uint4*)gsrc;
  uint4* dst = (uint4*)smem;
  for (int i = threadIdx.x; i < 1024; i += 256) dst[i] = src[i];
#endif
  __syncthreads();
}

// ------------------------------- utility kernels ---------------------------

__global__ void k_f32_to_bf16(const float* __restrict__ src, bf16* __restrict__ dst,
                              size_t n) {
  for (size_t i = (size_t)blockIdx.x * blockDim.x + threadIdx.x; i < n;
       i += (size_t)gridDim.x * blockDim.x)
    dst[i] = f2b(src[i]);
}

// dst[r*cols + c] = bf16(src[r*stride + off + c])
__global__ void k_cvt_rows(const float* __restrict__ src, size_t stride, size_t off,
                           bf16* __restrict__ dst, size_t rows, size_t cols) {
  size_t n = rows * cols;
  for (size_t i = (size_t)blockIdx.x * blockDim.x + threadIdx.x; i < n;
       i += (size_t)gridDim.x * blockDim.x) {
    size_t r = i / cols, c = i % cols;
    dst[i] = f2b(src[r * stride + off + c]);
  }
}

__global__ void k_zero(uint32_t* __restrict__ p, size_t n) {
  for (size_t i = (size_t)blockIdx.x * blockDim.x + threadIdx.x; i < n;
       i += (size_t)gridDim.x * blockDim.x)
    p[i] = 0u;
}

// ----------------------- phase 1: input projection GEMM --------------------
// pi[dir][t][b][g] = bias[g] + sum_k x[b][t][k] * Wi[g][k]   (bias folded in)
// 8192 blocks * 8 waves * grid.y=dir; wave = 16x64 tile, K=512.
// Software-pipelined: next k-step's 10 b128 loads issue before current WMMAs.
__global__ void k_input_gemm(const bf16* __restrict__ xbf,   // [2][B*T][512]
                             const bf16* __restrict__ WiB,   // [2][G][512]
                             const float* __restrict__ bias, // [2][G]
                             float* __restrict__ pi) {       // [2][T][B][G]
  const int dir = blockIdx.y;
  const int wave = (blockIdx.x << 3) | (threadIdx.x >> 5);
  const int l = threadIdx.x & 31;
  const int rt = wave >> 8;    // 0..255 row tile
  const int cbk = wave & 255;  // 0..255 col block (64 gate cols)
  const bf16* x = xbf + (size_t)dir * ((size_t)B_ * T_ * D_);
  const bf16* W = WiB + (size_t)dir * ((size_t)G_ * D_);
  const float* bs = bias + (size_t)dir * G_;
  float* po = pi + (size_t)dir * ((size_t)T_ * B_ * G_);
  const int m0 = rt * 16;
  const int c0 = cbk * 64;

  v8f acc[4];
#pragma unroll
  for (int j = 0; j < 4; j++) {
    float bv = bs[c0 + j * 16 + (l & 15)];
    acc[j] = (v8f){bv, bv, bv, bv, bv, bv, bv, bv};
  }
  // pipeline prologue
  v16bf a_cur = load_tile_x(x, m0, 0, l);
  v16bf b_cur[4];
#pragma unroll
  for (int j = 0; j < 4; j++) b_cur[j] = load_tile(W, D_, c0 + j * 16, 0, l);

#pragma unroll 2
  for (int k = 0; k < D_; k += 32) {
    // issue next iteration's loads (final-iter over-read stays in workspace)
    v16bf a_nxt = load_tile_x(x, m0, (k + 32) & (D_ - 1), l);
    v16bf b_nxt[4];
#pragma unroll
    for (int j = 0; j < 4; j++) b_nxt[j] = load_tile(W, D_, c0 + j * 16, k + 32, l);
    // consume current tiles
#pragma unroll
    for (int j = 0; j < 4; j++) acc[j] = WMMA_BF16(a_cur, b_cur[j], acc[j]);
    a_cur = a_nxt;
#pragma unroll
    for (int j = 0; j < 4; j++) b_cur[j] = b_nxt[j];
  }
#pragma unroll
  for (int j = 0; j < 4; j++) store_ctile(po, G_, m0, c0 + j * 16, acc[j], l);
}

// ----------------------- phase 2a: gates + cell update ---------------------
// One wave owns a 16x16 C-tile and computes all 4 gates (lane-local pointwise
// LSTM math). 8 waves/block share one A tile (16 rows of h = 16KB contiguous)
// staged to LDS via the Tensor Data Mover. 1024 waves = 128 blocks.
__global__ void k_step_cell(const bf16* __restrict__ h_bf,   // [2][32][512]
                            const bf16* __restrict__ WhB,    // [2][G][512]
                            const float* __restrict__ pi,    // [2][T][32][G]
                            const int* __restrict__ length,  // [32]
                            float* __restrict__ c_st,        // [2][32][4096]
                            bf16* __restrict__ a_bf,         // [2][32][4096]
                            int step) {
  __shared__ __align__(16) bf16 smemA[16 * H_];  // 16KB staged A tile
  const int wave = (blockIdx.x << 3) | (threadIdx.x >> 5);
  const int l = threadIdx.x & 31;
  const int dir = wave >> 9;          // uniform per block
  const int rest = wave & 511;
  const int mt = rest >> 8;           // uniform per block
  const int ct = rest & 255;          // per-wave 16-wide C column tile
  const int t_eff = dir ? (T_ - 1 - step) : step;
  const int m0 = mt * 16;
  const int n0 = ct * 16;
  const bf16* W = WhB + (size_t)dir * ((size_t)G_ * H_);
  const float* pit =
      pi + (size_t)dir * ((size_t)T_ * B_ * G_) + (size_t)t_eff * ((size_t)B_ * G_);

  // TDM: stage h rows [m0, m0+16) x 512 bf16 (one contiguous 16KB block)
  stage_16k_to_lds(h_bf + (size_t)dir * (B_ * H_) + (size_t)m0 * H_, smemA);

  v8f acc[4];  // i, f, g, o accumulators for the same (m, n) tile
#pragma unroll
  for (int g = 0; g < 4; g++) acc[g] = load_ctile(pit, G_, m0, g * C_ + n0, l);

  // software-pipelined K loop: A from LDS (ds_load_b128), B from global
  v16bf a_cur = load_tile(smemA, H_, 0, 0, l);
  v16bf b_cur[4];
#pragma unroll
  for (int g = 0; g < 4; g++) b_cur[g] = load_tile(W, H_, g * C_ + n0, 0, l);

#pragma unroll 2
  for (int k = 0; k < H_; k += 32) {
    v16bf a_nxt = load_tile(smemA, H_, 0, (k + 32) & (H_ - 1), l);
    v16bf b_nxt[4];
#pragma unroll
    for (int g = 0; g < 4; g++)
      b_nxt[g] = load_tile(W, H_, g * C_ + n0, k + 32, l);
#pragma unroll
    for (int g = 0; g < 4; g++) acc[g] = WMMA_BF16(a_cur, b_cur[g], acc[g]);
    a_cur = a_nxt;
#pragma unroll
    for (int g = 0; g < 4; g++) b_cur[g] = b_nxt[g];
  }

  const int n = n0 + (l & 15);
  float* cs = c_st + (size_t)dir * ((size_t)B_ * C_);
  bf16* as = a_bf + (size_t)dir * ((size_t)B_ * C_);
#pragma unroll
  for (int r = 0; r < 8; r++) {
    int b = m0 + r + ((l >> 4) << 3);
    float iv = 1.f / (1.f + __expf(-acc[0][r]));
    float fv = 1.f / (1.f + __expf(-acc[1][r]));
    float gv = tanhf(acc[2][r]);
    float ov = 1.f / (1.f + __expf(-acc[3][r]));
    size_t idx = (size_t)b * C_ + n;
    float cold = cs[idx];
    float cnew = fminf(fmaxf(iv * gv + fv * cold, -3.f), 3.f);
    bool valid = t_eff < length[b];
    cs[idx] = valid ? cnew : cold;
    as[idx] = f2b(ov * tanhf(cnew));
  }
}

// ----------------------- phase 2b: projection + output ---------------------
// h_new = clip(a @ Wp^T). Two independent WMMA chains over even/odd K-blocks
// plus software pipelining on the longest dependent chain (K=4096).
// 128 waves = 16 blocks.
__global__ void k_step_proj(const bf16* __restrict__ a_bf,   // [2][32][4096]
                            const bf16* __restrict__ WpB,    // [2][512][4096]
                            const int* __restrict__ length,  // [32]
                            float* __restrict__ h_st,        // [2][32][512]
                            bf16* __restrict__ h_bf,         // [2][32][512]
                            float* __restrict__ out_seq,     // [32][128][1024]
                            const float* __restrict__ res,   // nullable residual
                            int step) {
  const int wave = (blockIdx.x << 3) | (threadIdx.x >> 5);
  const int l = threadIdx.x & 31;
  const int dir = wave >> 6;
  const int rest = wave & 63;
  const int mt = rest >> 5;  // 0..1
  const int nt = rest & 31;  // 0..31 (H column tile)
  const int t_eff = dir ? (T_ - 1 - step) : step;
  const bf16* A = a_bf + (size_t)dir * ((size_t)B_ * C_);
  const bf16* W = WpB + (size_t)dir * ((size_t)H_ * C_);
  const int m0 = mt * 16, n0 = nt * 16;

  v8f acc0 = (v8f){0.f, 0.f, 0.f, 0.f, 0.f, 0.f, 0.f, 0.f};
  v8f acc1 = acc0;
  v16bf a0c = load_tile(A, C_, m0, 0, l);
  v16bf b0c = load_tile(W, C_, n0, 0, l);
  v16bf a1c = load_tile(A, C_, m0, 32, l);
  v16bf b1c = load_tile(W, C_, n0, 32, l);
#pragma unroll 2
  for (int k = 0; k < C_; k += 64) {
    v16bf a0n = load_tile(A, C_, m0, k + 64, l);
    v16bf b0n = load_tile(W, C_, n0, k + 64, l);
    v16bf a1n = load_tile(A, C_, m0, k + 96, l);
    v16bf b1n = load_tile(W, C_, n0, k + 96, l);
    acc0 = WMMA_BF16(a0c, b0c, acc0);
    acc1 = WMMA_BF16(a1c, b1c, acc1);
    a0c = a0n; b0c = b0n; a1c = a1n; b1c = b1n;
  }
  v8f acc = acc0 + acc1;

  const int n = n0 + (l & 15);
  float* hs = h_st + (size_t)dir * ((size_t)B_ * H_);
  bf16* hb = h_bf + (size_t)dir * ((size_t)B_ * H_);
#pragma unroll
  for (int r = 0; r < 8; r++) {
    int b = m0 + r + ((l >> 4) << 3);
    float hv = fminf(fmaxf(acc[r], -3.f), 3.f);
    bool valid = t_eff < length[b];
    size_t hidx = (size_t)b * H_ + n;
    float hold = hs[hidx];
    float h2 = valid ? hv : hold;
    hs[hidx] = h2;
    hb[hidx] = f2b(h2);
    float outv = valid ? hv : 0.f;
    size_t oidx = ((size_t)b * T_ + t_eff) * 1024 + (size_t)dir * 512 + n;
    if (res) outv += res[oidx];
    out_seq[oidx] = outv;
  }
}

// ----------------------- final-state gather --------------------------------
__global__ void k_finalize(const float* __restrict__ h_st,  // [2][32][512]
                           const float* __restrict__ c_st,  // [2][32][4096]
                           float* __restrict__ fin_h,       // [32][1024]
                           float* __restrict__ fin_c) {     // [32][8192]
  size_t nh = 2UL * B_ * H_;
  size_t nc = 2UL * B_ * C_;
  for (size_t i = (size_t)blockIdx.x * blockDim.x + threadIdx.x; i < nc;
       i += (size_t)gridDim.x * blockDim.x) {
    if (i < nh) {
      int dir = (int)(i / (B_ * H_));
      int rem = (int)(i % (B_ * H_));
      int b = rem / H_, n = rem % H_;
      fin_h[(size_t)b * (2 * H_) + (size_t)dir * H_ + n] = h_st[i];
    }
    int dir = (int)(i / ((size_t)B_ * C_));
    int rem = (int)(i % ((size_t)B_ * C_));
    int b = rem / C_, n = rem % C_;
    fin_c[(size_t)b * (2 * C_) + (size_t)dir * C_ + n] = c_st[i];
  }
}

// ---------------------------------------------------------------------------

extern "C" void kernel_launch(void* const* d_in, const int* in_sizes, int n_in,
                              void* d_out, int out_size, void* d_ws, size_t ws_size,
                              hipStream_t stream) {
  (void)in_sizes; (void)n_in; (void)out_size; (void)ws_size;
  const float* in_x = (const float*)d_in[0];          // [32][128][512]
  const int* length = (const int*)d_in[1];            // [32]
  const float* Wi = (const float*)d_in[2];            // [2][2][16384][512]
  const float* Wh = (const float*)d_in[3];            // [2][2][16384][512]
  const float* bias = (const float*)d_in[4];          // [2][2][16384]
  const float* Wp = (const float*)d_in[5];            // [2][2][512][4096]
  float* out = (float*)d_out;

  // ---- workspace layout (256B aligned slabs) ----
  size_t o = 0;
  auto alloc = [&](size_t bytes) {
    size_t r = o;
    o = (o + bytes + 255) & ~(size_t)255;
    return r;
  };
  char* ws = (char*)d_ws;
  size_t oWiB = alloc((size_t)L_ * 2 * G_ * D_ * sizeof(bf16));   // 67.1 MB
  size_t oWhB = alloc((size_t)L_ * 2 * G_ * H_ * sizeof(bf16));   // 67.1 MB
  size_t oWpB = alloc((size_t)L_ * 2 * H_ * C_ * sizeof(bf16));   // 16.8 MB
  size_t oXbf = alloc((size_t)2 * B_ * T_ * D_ * sizeof(bf16));   //  8.4 MB
  size_t oPi  = alloc((size_t)2 * T_ * B_ * G_ * sizeof(float));  // 537  MB
  size_t oAbf = alloc((size_t)2 * B_ * C_ * sizeof(bf16));
  size_t oH   = alloc((size_t)2 * B_ * H_ * sizeof(float));
  size_t oC   = alloc((size_t)2 * B_ * C_ * sizeof(float));
  size_t oHbf = alloc((size_t)2 * B_ * H_ * sizeof(bf16));
  size_t oPad = alloc((size_t)4096);  // over-read slack for pipelined tails
  (void)oPad;
  size_t stateBytes = (oHbf + (size_t)2 * B_ * H_ * sizeof(bf16)) - oH;

  bf16* WiB = (bf16*)(ws + oWiB);
  bf16* WhB = (bf16*)(ws + oWhB);
  bf16* WpB = (bf16*)(ws + oWpB);
  bf16* Xbf = (bf16*)(ws + oXbf);
  float* Pi = (float*)(ws + oPi);
  bf16* Abf = (bf16*)(ws + oAbf);
  float* Hst = (float*)(ws + oH);
  float* Cst = (float*)(ws + oC);
  bf16* Hbf = (bf16*)(ws + oHbf);

  const dim3 blk(256);
  // ---- one-time weight conversion to bf16 ----
  k_f32_to_bf16<<<4096, blk, 0, stream>>>(Wi, WiB, (size_t)L_ * 2 * G_ * D_);
  k_f32_to_bf16<<<4096, blk, 0, stream>>>(Wh, WhB, (size_t)L_ * 2 * G_ * H_);
  k_f32_to_bf16<<<4096, blk, 0, stream>>>(Wp, WpB, (size_t)L_ * 2 * H_ * C_);

  float* finH = out + (size_t)L_ * B_ * T_ * 1024;  // [L][32][1024]
  float* finC = finH + (size_t)L_ * B_ * 2 * H_;    // [L][32][8192]

  for (int lyr = 0; lyr < L_; ++lyr) {
    // layer inputs -> bf16 (layer 0: raw x for both dirs; layer 1: prev seq out)
    for (int dir = 0; dir < 2; ++dir) {
      bf16* dst = Xbf + (size_t)dir * ((size_t)B_ * T_ * D_);
      if (lyr == 0)
        k_cvt_rows<<<4096, blk, 0, stream>>>(in_x, (size_t)D_, 0, dst,
                                             (size_t)B_ * T_, (size_t)D_);
      else
        k_cvt_rows<<<4096, blk, 0, stream>>>(out, (size_t)1024, (size_t)dir * 512,
                                             dst, (size_t)B_ * T_, (size_t)D_);
    }
    // zero h/c state (fp32 + bf16 mirrors)
    k_zero<<<1024, blk, 0, stream>>>((uint32_t*)(ws + oH), stateBytes / 4);

    // input projection GEMM, both dirs, bias folded
    const bf16* WiL = WiB + (size_t)lyr * 2 * G_ * D_;
    const float* bL = bias + (size_t)lyr * 2 * G_;
    k_input_gemm<<<dim3(8192, 2), blk, 0, stream>>>(Xbf, WiL, bL, Pi);

    // recurrence: 128 dependent steps, fwd+bwd fused per launch
    const bf16* WhL = WhB + (size_t)lyr * 2 * G_ * H_;
    const bf16* WpL = WpB + (size_t)lyr * 2 * H_ * C_;
    float* seqL = out + (size_t)lyr * B_ * T_ * 1024;
    const float* res = (lyr == 0) ? nullptr : out;  // residual = layer-0 seq out
    for (int s = 0; s < T_; ++s) {
      k_step_cell<<<128, blk, 0, stream>>>(Hbf, WhL, Pi, length, Cst, Abf, s);
      k_step_proj<<<16, blk, 0, stream>>>(Abf, WpL, length, Hst, Hbf, seqL, res, s);
    }
    k_finalize<<<1024, blk, 0, stream>>>(Hst, Cst, finH + (size_t)lyr * B_ * 2 * H_,
                                         finC + (size_t)lyr * B_ * 2 * C_);
  }
}
